// node_in_edge_loss_60026462929136
// MI455X (gfx1250) — matrix operand order repaired and checked by reading, not compiled, not run on hardware
//
#include <hip/hip_runtime.h>
#include <hip/hip_bf16.h>

typedef __attribute__((ext_vector_type(16))) __bf16 v16bf;
typedef __attribute__((ext_vector_type(8)))  float  v8f;

#define NUM_NODES 100000
#define NUM_EDGES 50000
#define EMB 128
#define HID 128
#define BATCH 65536
#define BN_EPS 1e-5f

#define WT_ROW 264   // 256 ushorts + 8 pad -> 132 dwords, bank-conflict-free B reads

// ---------------- helpers ----------------

__device__ __forceinline__ unsigned int pack2_bf16(float a, float b) {
    // round-to-nearest-even f32 -> bf16, pack two into one dword
    unsigned int ua = __float_as_uint(a);
    unsigned int ub = __float_as_uint(b);
    ua = (ua + 0x7FFFu + ((ua >> 16) & 1u)) >> 16;
    ub = (ub + 0x7FFFu + ((ub >> 16) & 1u)) >> 16;
    return (ua & 0xFFFFu) | (ub << 16);
}

union BFrag16 { v16bf v; unsigned int u[8]; };
union F8      { v8f   v; float        f[8]; };

// ---------------- kernel 0: W1 (256x128 f32, row-major K x N) -> Wt (128x256 bf16, N-major) ----

__global__ void giel_prep_wt(const float* __restrict__ W1, unsigned short* __restrict__ Wt) {
    const int n = blockIdx.x;     // 0..127 output feature
    const int k = threadIdx.x;    // 0..255 input feature
    unsigned int u = __float_as_uint(W1[k * HID + n]);
    u = (u + 0x7FFFu + ((u >> 16) & 1u)) >> 16;
    Wt[n * 256 + k] = (unsigned short)u;
}

// ---------------- kernel 1: gather + GEMM1 (bf16 WMMA) + bias + ReLU + BN partial stats ----

__global__ __launch_bounds__(256) void giel_gemm1(
    const float* __restrict__ nodeEmb, const float* __restrict__ edgeEmb,
    const int*   __restrict__ nodeIdx, const int*   __restrict__ edgeIdx,
    const float* __restrict__ b1,
    const unsigned short* __restrict__ Wt,     // [128][256] bf16
    float* __restrict__ H,                     // [BATCH][HID]
    float* __restrict__ gsum, float* __restrict__ gsq)  // [gridDim][128]
{
    __shared__ float lsum[8][128];
    __shared__ float lsq[8][128];
    extern __shared__ unsigned short ldsW[];   // [128][WT_ROW] bf16 weight tile

    const int tid  = threadIdx.x;
    const int lane = tid & 31;
    const int wave = tid >> 5;          // 0..7
    const int l15  = lane & 15;
    const int hi   = lane >> 4;         // 0 or 1 (lane half)

    // ---- stage Wt (64 KB bf16) into LDS once per WG, padded rows ----
    {
        const uint4* src = (const uint4*)Wt;       // 16-byte chunks, 2048 total
        #pragma unroll
        for (int i = 0; i < 8; ++i) {
            const int cid = i * 256 + tid;         // chunk id
            const int row = cid >> 4;              // 16 chunks per 256-elem row
            const int col = (cid & 15) * 8;        // ushort column
            uint4 d = src[cid];
            *(uint4*)&ldsW[row * WT_ROW + col] = d;
        }
    }

    const int rowBase = (blockIdx.x * 8 + wave) * 16;   // 16 rows per wave
    const int myRow   = rowBase + l15;                  // both lane halves share a row

    const float* nrow = nodeEmb + (size_t)nodeIdx[myRow] * EMB;
    const float* erow = edgeEmb + (size_t)edgeIdx[myRow] * EMB;

    F8 acc[8];
    #pragma unroll
    for (int n = 0; n < 8; ++n)
        #pragma unroll
        for (int j = 0; j < 8; ++j) acc[n].f[j] = 0.0f;

    // A-side software pipeline: gathered f32 data for the *next* K-step is in
    // flight while the current 8-WMMA burst executes.
    float4 fa0, fa1, fa2, fa3;
    auto loadA = [&](int ks) {
        const int k0 = ks * 32;
        // concat: K 0..127 from node table, 128..255 from edge table
        const float* src = (k0 < EMB) ? (nrow + k0) : (erow + (k0 - EMB));
        const float* p   = src + hi * 8;
        fa0 = *(const float4*)(p);
        fa1 = *(const float4*)(p + 4);
        fa2 = *(const float4*)(p + 16);
        fa3 = *(const float4*)(p + 20);
    };
    loadA(0);

    __syncthreads();   // LDS weight tile ready

    #pragma unroll 1
    for (int ks = 0; ks < 8; ++ks) {
        // A fragment, 16-bit 16x32 layout:
        //  lanes 0-15 : elems 0-7 = K k0+0..7,  elems 8-15 = K k0+16..23
        //  lanes 16-31: elems 0-7 = K k0+8..15, elems 8-15 = K k0+24..31
        BFrag16 a;
        a.u[0] = pack2_bf16(fa0.x, fa0.y); a.u[1] = pack2_bf16(fa0.z, fa0.w);
        a.u[2] = pack2_bf16(fa1.x, fa1.y); a.u[3] = pack2_bf16(fa1.z, fa1.w);
        a.u[4] = pack2_bf16(fa2.x, fa2.y); a.u[5] = pack2_bf16(fa2.z, fa2.w);
        a.u[6] = pack2_bf16(fa3.x, fa3.y); a.u[7] = pack2_bf16(fa3.z, fa3.w);

        // B fragments (all 8 N-tiles) from LDS: column N = l15,
        // lanes 0-15 hold K k0+0..7,16..23, lanes 16-31 hold K k0+8..15,24..31
        const int kb = ks * 32 + hi * 8;
        BFrag16 b[8];
        #pragma unroll
        for (int n = 0; n < 8; ++n) {
            const unsigned short* wp = &ldsW[(n * 16 + l15) * WT_ROW + kb];
            uint4 w0 = *(const uint4*)(wp);
            uint4 w1 = *(const uint4*)(wp + 16);
            b[n].u[0] = w0.x; b[n].u[1] = w0.y; b[n].u[2] = w0.z; b[n].u[3] = w0.w;
            b[n].u[4] = w1.x; b[n].u[5] = w1.y; b[n].u[6] = w1.z; b[n].u[7] = w1.w;
        }

        if (ks < 7) loadA(ks + 1);   // issue next gather before the WMMA burst

        #pragma unroll
        for (int n = 0; n < 8; ++n) {
            acc[n].v = __builtin_amdgcn_wmma_f32_16x16x32_bf16(
                false, a.v, false, b[n].v, (short)0, acc[n].v, false, false);
        }
    }

    // C layout: VGPR r -> row M = r (+8 for lane half 1), col N = l15
    float csum[8], csq[8];
    #pragma unroll
    for (int n = 0; n < 8; ++n) {
        const int col  = n * 16 + l15;
        const float bv = b1[col];
        float s = 0.0f, q = 0.0f;
        #pragma unroll
        for (int r = 0; r < 8; ++r) {
            const int row = rowBase + r + hi * 8;
            float v = acc[n].f[r] + bv;
            v = v > 0.0f ? v : 0.0f;
            H[(size_t)row * HID + col] = v;
            s += v; q += v * v;
        }
        csum[n] = s; csq[n] = q;
    }

    // fold lane halves (same columns, different rows), stage per-wave partials in LDS
    #pragma unroll
    for (int n = 0; n < 8; ++n) {
        float s2 = csum[n] + __shfl_xor(csum[n], 16, 32);
        float q2 = csq[n]  + __shfl_xor(csq[n],  16, 32);
        if (hi == 0) { lsum[wave][n * 16 + l15] = s2; lsq[wave][n * 16 + l15] = q2; }
    }
    __syncthreads();

    if (tid < 128) {
        float s = 0.0f, q = 0.0f;
        #pragma unroll
        for (int w = 0; w < 8; ++w) { s += lsum[w][tid]; q += lsq[w][tid]; }
        gsum[blockIdx.x * 128 + tid] = s;
        gsq [blockIdx.x * 128 + tid] = q;
    }
}

// ---------------- kernel 2: finalize BN stats -> per-feature scale/shift ----

__global__ void giel_bn_stats(const float* __restrict__ gsum, const float* __restrict__ gsq,
                              const float* __restrict__ gamma, const float* __restrict__ beta,
                              float* __restrict__ scale, float* __restrict__ shift,
                              int nparts)
{
    const int t = threadIdx.x;   // 0..127
    float s = 0.0f, q = 0.0f;
    for (int w = 0; w < nparts; ++w) { s += gsum[w * 128 + t]; q += gsq[w * 128 + t]; }
    const float mean = s * (1.0f / BATCH);
    const float var  = q * (1.0f / BATCH) - mean * mean;      // biased batch variance
    const float inv  = rsqrtf(var + BN_EPS);
    const float sc   = gamma[t] * inv;
    scale[t] = sc;
    shift[t] = beta[t] - mean * sc;
}

// ---------------- kernel 3: normalize + head GEMM (128->2) + log-softmax NLL, block partial ----

__global__ __launch_bounds__(256) void giel_head_loss(
    const float* __restrict__ H, const float* __restrict__ scale,
    const float* __restrict__ shift, const float* __restrict__ W2,
    const float* __restrict__ b2, const int* __restrict__ labels,
    float* __restrict__ ploss)
{
    __shared__ float sW0[128], sW1[128], sSc[128], sSh[128];
    __shared__ float red[256];
    const int tid = threadIdx.x;
    if (tid < 128) {
        sW0[tid] = W2[tid * 2 + 0];
        sW1[tid] = W2[tid * 2 + 1];
        sSc[tid] = scale[tid];
        sSh[tid] = shift[tid];
    }
    __syncthreads();

    const int row = blockIdx.x * 256 + tid;
    const float* hp = H + (size_t)row * HID;
    float a0 = b2[0], a1 = b2[1];
    #pragma unroll 4
    for (int j = 0; j < HID; j += 4) {
        float4 hv = *(const float4*)(hp + j);
        float x0 = hv.x * sSc[j + 0] + sSh[j + 0]; a0 += x0 * sW0[j + 0]; a1 += x0 * sW1[j + 0];
        float x1 = hv.y * sSc[j + 1] + sSh[j + 1]; a0 += x1 * sW0[j + 1]; a1 += x1 * sW1[j + 1];
        float x2 = hv.z * sSc[j + 2] + sSh[j + 2]; a0 += x2 * sW0[j + 2]; a1 += x2 * sW1[j + 2];
        float x3 = hv.w * sSc[j + 3] + sSh[j + 3]; a0 += x3 * sW0[j + 3]; a1 += x3 * sW1[j + 3];
    }
    const float m   = fmaxf(a0, a1);
    const float lse = m + logf(expf(a0 - m) + expf(a1 - m));
    const float zl  = labels[row] ? a1 : a0;
    red[tid] = lse - zl;
    __syncthreads();
    #pragma unroll
    for (int s = 128; s > 0; s >>= 1) {
        if (tid < s) red[tid] += red[tid + s];
        __syncthreads();
    }
    if (tid == 0) ploss[blockIdx.x] = red[0];
}

// ---------------- kernel 4: final mean ----

__global__ void giel_final(const float* __restrict__ ploss, float* __restrict__ out) {
    __shared__ float red[256];
    const int tid = threadIdx.x;
    red[tid] = ploss[tid];
    __syncthreads();
    #pragma unroll
    for (int s = 128; s > 0; s >>= 1) {
        if (tid < s) red[tid] += red[tid + s];
        __syncthreads();
    }
    if (tid == 0) out[0] = red[0] * (1.0f / BATCH);
}

// ---------------- launch ----------------

extern "C" void kernel_launch(void* const* d_in, const int* in_sizes, int n_in,
                              void* d_out, int out_size, void* d_ws, size_t ws_size,
                              hipStream_t stream) {
    const float* nodeEmb = (const float*)d_in[0];
    const float* edgeEmb = (const float*)d_in[1];
    const int*   nodeIdx = (const int*)  d_in[2];
    const int*   edgeIdx = (const int*)  d_in[3];
    const int*   labels  = (const int*)  d_in[4];
    const float* W1      = (const float*)d_in[5];
    const float* b1      = (const float*)d_in[6];
    const float* gamma   = (const float*)d_in[7];
    const float* beta    = (const float*)d_in[8];
    const float* W2      = (const float*)d_in[9];
    const float* b2      = (const float*)d_in[10];
    float* out = (float*)d_out;

    constexpr int    NWG1   = 512;                               // gemm1 workgroups
    constexpr size_t H_OFF  = 0;
    constexpr size_t WT_OFF = (size_t)BATCH * HID * sizeof(float);       // 33.5 MB
    constexpr size_t GS_OFF = WT_OFF + (size_t)HID * 256 * 2;            // bf16 Wt
    constexpr size_t GQ_OFF = GS_OFF + (size_t)NWG1 * 128 * sizeof(float);
    constexpr size_t SC_OFF = GQ_OFF + (size_t)NWG1 * 128 * sizeof(float);
    constexpr size_t SH_OFF = SC_OFF + 128 * sizeof(float);
    constexpr size_t PL_OFF = SH_OFF + 128 * sizeof(float);

    char* ws = (char*)d_ws;
    float*          H     = (float*)(ws + H_OFF);
    unsigned short* Wt    = (unsigned short*)(ws + WT_OFF);
    float*          gsum  = (float*)(ws + GS_OFF);
    float*          gsq   = (float*)(ws + GQ_OFF);
    float*          scale = (float*)(ws + SC_OFF);
    float*          shift = (float*)(ws + SH_OFF);
    float*          ploss = (float*)(ws + PL_OFF);

    const size_t ldsBytes = (size_t)HID * WT_ROW * sizeof(unsigned short);  // 66 KB

    giel_prep_wt <<<dim3(HID),  dim3(256), 0, stream>>>(W1, Wt);
    giel_gemm1   <<<dim3(NWG1), dim3(256), ldsBytes, stream>>>(nodeEmb, edgeEmb, nodeIdx, edgeIdx,
                                                               b1, Wt, H, gsum, gsq);
    giel_bn_stats<<<dim3(1),    dim3(128), 0, stream>>>(gsum, gsq, gamma, beta, scale, shift, NWG1);
    giel_head_loss<<<dim3(BATCH / 256), dim3(256), 0, stream>>>(H, scale, shift, W2, b2, labels, ploss);
    giel_final   <<<dim3(1),    dim3(256), 0, stream>>>(ploss, out);
}
